// DSARingIndexer_40372692583236
// MI455X (gfx1250) — matrix operand order repaired and checked by reading, not compiled, not run on hardware
//
#include <hip/hip_runtime.h>
#include <hip/hip_bf16.h>

// ---- problem dims (from reference setup_inputs) ----
#define BSZ 2
#define SL  4096
#define SG  8192
#define QD  1536
#define KVR 512
#define DD  128
#define TEMP 11.313708498984761f   // sqrt(128)
#define NEG_INF (-3.0e38f)

typedef __attribute__((ext_vector_type(16))) __bf16 v16bf;
typedef __attribute__((ext_vector_type(8)))  __bf16 v8bf;
typedef __attribute__((ext_vector_type(8)))  float  v8f;

// ---- workspace layout (bytes, 256-aligned) ----
#define OFF_QC 0                                  // bf16 q_compressed  (2*4096*1536)
#define OFF_KV (OFF_QC + (size_t)BSZ*SL*QD*2)     // bf16 all_kv        (2*8192*512)
#define OFF_WQ (OFF_KV + (size_t)BSZ*SG*KVR*2)    // bf16 Wq            (128*1536)
#define OFF_WK (OFF_WQ + (size_t)DD*QD*2)         // bf16 Wk            (128*512)
#define OFF_QI (OFF_WK + (size_t)DD*KVR*2)        // bf16 q_idx         (2*4096*128)
#define OFF_KI (OFF_QI + (size_t)BSZ*SL*DD*2)     // bf16 k_idx         (2*8192*128)
#define OFF_SC (OFF_KI + (size_t)BSZ*SG*DD*2)     // f32 scores         (2*4096*8192)

static __device__ __forceinline__ v8bf load8bf(const unsigned short* p) {
  union { uint4 u; v8bf v; } t;
  t.u = *(const uint4*)p;
  return t.v;
}
static __device__ __forceinline__ v16bf cat16(v8bf lo, v8bf hi) {
  return __builtin_shufflevector(lo, hi, 0,1,2,3,4,5,6,7,8,9,10,11,12,13,14,15);
}
static __device__ __forceinline__ unsigned short f2bf(float f) {
  unsigned x = __float_as_uint(f);
  unsigned r = x + 0x7FFFu + ((x >> 16) & 1u);   // round-to-nearest-even
  return (unsigned short)(r >> 16);
}
static __device__ __forceinline__ unsigned fkey(float f) {
  unsigned x = __float_as_uint(f);               // monotonic total order on floats
  return (x & 0x80000000u) ? ~x : (x | 0x80000000u);
}

// ---- Tensor Data Mover: contiguous 4KB global -> LDS copy (one D#) ----
#if __has_builtin(__builtin_amdgcn_tensor_load_to_lds) && \
    __has_builtin(__builtin_amdgcn_s_wait_tensorcnt)
#define HAVE_TDM 1
typedef unsigned int u32x4 __attribute__((ext_vector_type(4)));
typedef int          i32x4 __attribute__((ext_vector_type(4)));
typedef int          i32x8 __attribute__((ext_vector_type(8)));

static __device__ __forceinline__ void tdm_copy_4k(unsigned lds_addr, const void* gsrc) {
  unsigned long long ga = (unsigned long long)gsrc;
  // force wave-uniform (TDM descriptor lives in SGPRs)
  unsigned l0 = (unsigned)__builtin_amdgcn_readfirstlane((int)lds_addr);
  unsigned a0 = (unsigned)__builtin_amdgcn_readfirstlane((int)(ga & 0xFFFFFFFFu));
  unsigned a1 = (unsigned)__builtin_amdgcn_readfirstlane((int)(ga >> 32));

  u32x4 g0;
  g0[0] = 1u;                                    // count=1: valid descriptor
  g0[1] = l0;                                    // lds_addr (bytes)
  g0[2] = a0;                                    // global_addr[31:0]
  g0[3] = (a1 & 0x01FFFFFFu) | (2u << 30);       // global_addr[56:32] | type=2
  i32x8 g1;
  g1[0] = (int)(3u << 16);                       // data_size=8B, no multicast/pad
  g1[1] = (int)(512u << 16);                     // tensor_dim0 = 512 (lo16 @ [31:16])
  g1[2] = (int)(1u << 16);                       // tensor_dim0 hi=0 | tensor_dim1 = 1
  g1[3] = (int)(512u << 16);                     // tensor_dim1 hi=0 | tile_dim0 = 512
  g1[4] = 1;                                     // tile_dim1 = 1, tile_dim2 = 0
  g1[5] = 512;                                   // tensor_dim0_stride = 512
  g1[6] = 0;
  g1[7] = 0;
  i32x4 z4 = {0, 0, 0, 0};
#if __clang_major__ >= 23
  i32x8 z8 = {0, 0, 0, 0, 0, 0, 0, 0};
  __builtin_amdgcn_tensor_load_to_lds(g0, g1, z4, z4, z8, 0);
#else
  __builtin_amdgcn_tensor_load_to_lds(g0, g1, z4, z4, 0);
#endif
}
#endif

// ---------------- kernel 0: f32 -> bf16 ----------------
__global__ void dsa_cvt_bf16(const float* __restrict__ in, unsigned short* __restrict__ out, int n) {
  int i = blockIdx.x * blockDim.x + threadIdx.x;
  if (i < n) out[i] = f2bf(in[i]);
}

// ---------------- kernel 1: k projection + layernorm ----------------
__global__ void __launch_bounds__(32) dsa_kproj_ln(
    const unsigned short* __restrict__ kv, const unsigned short* __restrict__ wk,
    const float* __restrict__ bk, const float* __restrict__ gamma,
    const float* __restrict__ beta, unsigned short* __restrict__ kidx) {
  __shared__ float sm[16 * DD];
  const int lane = threadIdx.x & 31;
  const int m = lane & 15, h = lane >> 4;
  const long rowbase = (long)blockIdx.x * 16;

  v8f acc[8];
  const v8f vz = {0.f,0.f,0.f,0.f,0.f,0.f,0.f,0.f};
  #pragma unroll
  for (int t = 0; t < 8; ++t) acc[t] = vz;

  for (int ks = 0; ks < KVR / 32; ++ks) {
    const int kb = ks * 32;
    const unsigned short* ar = kv + (rowbase + m) * KVR + kb;
    v16bf A = cat16(load8bf(ar + 16*h), load8bf(ar + 16*h + 8));
    #pragma unroll
    for (int t = 0; t < 8; ++t) {
      const unsigned short* br = wk + (size_t)(t*16 + m) * KVR + kb + 16*h;
      v16bf B = cat16(load8bf(br), load8bf(br + 8));
      acc[t] = __builtin_amdgcn_wmma_f32_16x16x32_bf16(false, A, false, B,
                                                       (short)0, acc[t], false, false);
    }
  }
  #pragma unroll
  for (int t = 0; t < 8; ++t)
    #pragma unroll
    for (int r = 0; r < 8; ++r)
      sm[(8*h + r) * DD + t*16 + m] = acc[t][r] + bk[t*16 + m];
  __syncthreads();

  if (lane < 16) {
    float mu = 0.f;
    for (int j = 0; j < DD; ++j) mu += sm[lane*DD + j];
    mu *= (1.0f / DD);
    float var = 0.f;
    for (int j = 0; j < DD; ++j) { float d = sm[lane*DD + j] - mu; var += d*d; }
    var *= (1.0f / DD);
    float rstd = rsqrtf(var + 1e-6f);
    for (int j = 0; j < DD; ++j) {
      float x = (sm[lane*DD + j] - mu) * rstd * gamma[j] + beta[j];
      kidx[(rowbase + lane) * DD + j] = f2bf(x);
    }
  }
}

// ---------------- kernel 2: q projection (+bias, /temperature) ----------------
__global__ void __launch_bounds__(32) dsa_qproj(
    const unsigned short* __restrict__ qc, const unsigned short* __restrict__ wq,
    const float* __restrict__ bq, unsigned short* __restrict__ qidx) {
  const int lane = threadIdx.x & 31;
  const int m = lane & 15, h = lane >> 4;
  const long rowbase = (long)blockIdx.x * 16;

  v8f acc[8];
  const v8f vz = {0.f,0.f,0.f,0.f,0.f,0.f,0.f,0.f};
  #pragma unroll
  for (int t = 0; t < 8; ++t) acc[t] = vz;

  for (int ks = 0; ks < QD / 32; ++ks) {
    const int kb = ks * 32;
    const unsigned short* ar = qc + (rowbase + m) * QD + kb;
    v16bf A = cat16(load8bf(ar + 16*h), load8bf(ar + 16*h + 8));
    #pragma unroll
    for (int t = 0; t < 8; ++t) {
      const unsigned short* br = wq + (size_t)(t*16 + m) * QD + kb + 16*h;
      v16bf B = cat16(load8bf(br), load8bf(br + 8));
      acc[t] = __builtin_amdgcn_wmma_f32_16x16x32_bf16(false, A, false, B,
                                                       (short)0, acc[t], false, false);
    }
  }
  const float invT = 1.0f / TEMP;   // fold temperature into q_idx
  #pragma unroll
  for (int t = 0; t < 8; ++t)
    #pragma unroll
    for (int r = 0; r < 8; ++r) {
      float v = (acc[t][r] + bq[t*16 + m]) * invT;
      qidx[(rowbase + 8*h + r) * DD + t*16 + m] = f2bf(v);
    }
}

// ---------------- kernel 3: scores (TDM-staged WMMA) + radix top-k + softmax ----------------
// 512 threads = 16 waves per 16-query tile. Phase 1: each wave scores 512 keys
// for all 16 queries, staging k_idx tiles LDS-side via the Tensor Data Mover
// with a 2-deep pipeline. Phase 2: wave w does top-k for query row w.
__global__ void __launch_bounds__(512) dsa_scores_topk(
    const unsigned short* __restrict__ qidx, const unsigned short* __restrict__ kidx,
    float* __restrict__ scoresWS, const int* __restrict__ cp_rank_p, int K,
    int* __restrict__ outIdx, float* __restrict__ outScore) {
  __shared__ unsigned hist[16][256];
  __shared__ unsigned selB[16], selK[16], cntG[16], cntT[16];
#ifdef HAVE_TDM
  extern __shared__ __align__(16) char dynlds[];   // 16 waves x 2 x 4KB
#endif

  const int tid = threadIdx.x;
  const int wave = tid >> 5, lane = tid & 31;
  const int n = lane & 15, h = lane >> 4;
  const int b  = blockIdx.x / (SL / 16);
  const int qt = blockIdx.x % (SL / 16);
  const int cp = cp_rank_p[0];
  const long qrow0 = (long)b * SL + qt * 16;

  // ---- phase 1: score BMM, causal mask, spill to workspace ----
  v16bf A[4];
  #pragma unroll
  for (int kk = 0; kk < 4; ++kk) {
    const unsigned short* ar = qidx + (qrow0 + n) * DD + kk * 32;
    A[kk] = cat16(load8bf(ar + 16*h), load8bf(ar + 16*h + 8));
  }
  const v8f vz = {0.f,0.f,0.f,0.f,0.f,0.f,0.f,0.f};

#ifdef HAVE_TDM
  char* mybuf = dynlds + (size_t)wave * 8192;      // double buffer for this wave
  {
    const long krow0 = (long)b * SG + wave * 512;  // nt = 0 tile (16 keys x 256B)
    tdm_copy_4k((unsigned)(size_t)(void*)mybuf, kidx + krow0 * DD);
  }
  for (int nt = 0; nt < 32; ++nt) {
    if (nt + 1 < 32) {
      const long krow1 = (long)b * SG + wave * 512 + (nt + 1) * 16;
      tdm_copy_4k((unsigned)(size_t)(void*)(mybuf + ((nt + 1) & 1) * 4096),
                  kidx + krow1 * DD);
      __builtin_amdgcn_s_wait_tensorcnt(1);        // tile nt done, nt+1 in flight
    } else {
      __builtin_amdgcn_s_wait_tensorcnt(0);        // last tile done
    }
    const unsigned short* bt = (const unsigned short*)(mybuf + (nt & 1) * 4096);
    const int keybase = wave * 512 + nt * 16;
    v8f c = vz;
    #pragma unroll
    for (int kk = 0; kk < 4; ++kk) {
      const unsigned short* br = bt + n * DD + kk * 32 + 16*h;   // ds_load_b128 x2
      v16bf B = cat16(load8bf(br), load8bf(br + 8));
      c = __builtin_amdgcn_wmma_f32_16x16x32_bf16(false, A[kk], false, B,
                                                  (short)0, c, false, false);
    }
    const int kvpos = keybase + n;
    #pragma unroll
    for (int r = 0; r < 8; ++r) {
      const int M = 8*h + r;
      const int qpos = qt*16 + M + cp * SL;
      float s = (kvpos <= qpos) ? c[r] : NEG_INF;
      scoresWS[(size_t)(qrow0 + M) * SG + kvpos] = s;
    }
  }
#else
  for (int nt = 0; nt < 32; ++nt) {
    const int keybase = wave * 512 + nt * 16;
    const long krow0 = (long)b * SG + keybase;
    v8f c = vz;
    #pragma unroll
    for (int kk = 0; kk < 4; ++kk) {
      const unsigned short* br = kidx + (krow0 + n) * DD + kk * 32 + 16*h;
      v16bf B = cat16(load8bf(br), load8bf(br + 8));
      c = __builtin_amdgcn_wmma_f32_16x16x32_bf16(false, A[kk], false, B,
                                                  (short)0, c, false, false);
    }
    const int kvpos = keybase + n;
    #pragma unroll
    for (int r = 0; r < 8; ++r) {
      const int M = 8*h + r;
      const int qpos = qt*16 + M + cp * SL;
      float s = (kvpos <= qpos) ? c[r] : NEG_INF;
      scoresWS[(size_t)(qrow0 + M) * SG + kvpos] = s;
    }
  }
#endif
  __threadfence();
  __syncthreads();

  // ---- phase 2: per-wave top-K on row `wave` ----
  const long rowIdx = qrow0 + wave;
  const float* rs = scoresWS + (size_t)rowIdx * SG;

  float mx = NEG_INF;
  for (int j = lane; j < SG; j += 32) {
    __builtin_prefetch(rs + j + 512, 0, 0);
    mx = fmaxf(mx, rs[j]);
  }
  #pragma unroll
  for (int off = 16; off > 0; off >>= 1) mx = fmaxf(mx, __shfl_xor(mx, off, 32));

  unsigned prefix = 0, pmask = 0, krem = (unsigned)K;
  for (int p = 0; p < 4; ++p) {
    const int shift = 24 - 8*p;
    for (int i = lane; i < 256; i += 32) hist[wave][i] = 0;
    __syncthreads();
    for (int j = lane; j < SG; j += 32) {
      unsigned u = fkey(rs[j]);
      if ((u & pmask) == prefix) atomicAdd(&hist[wave][(u >> shift) & 255u], 1u);
    }
    __syncthreads();
    if (lane == 0) {
      unsigned cum = 0;
      for (int bb = 255; bb >= 0; --bb) {
        unsigned cbin = hist[wave][bb];
        if (cum + cbin >= krem) { selB[wave] = (unsigned)bb; selK[wave] = krem - cum; break; }
        cum += cbin;
      }
      cntG[wave] = 0; cntT[wave] = 0;
    }
    __syncthreads();
    prefix |= selB[wave] << shift;
    pmask  |= 0xFFu << shift;
    krem = selK[wave];
  }
  const unsigned T = prefix, kTie = krem;
  const unsigned nG = (unsigned)K - kTie;
  const size_t base = (size_t)rowIdx * (size_t)K;

  float lsum = 0.f;
  for (int j = lane; j < SG; j += 32) {
    float s = rs[j];
    unsigned u = fkey(s);
    if (u > T) {
      unsigned pos = atomicAdd(&cntG[wave], 1u);
      outIdx[base + pos] = j; outScore[base + pos] = s;
      lsum += __expf(s - mx);
    } else if (u == T) {
      unsigned pos = atomicAdd(&cntT[wave], 1u);
      if (pos < kTie) {
        outIdx[base + nG + pos] = j; outScore[base + nG + pos] = s;
        lsum += __expf(s - mx);
      }
    }
  }
  #pragma unroll
  for (int off = 16; off > 0; off >>= 1) lsum += __shfl_xor(lsum, off, 32);
  asm volatile("s_wait_storecnt 0" ::: "memory");   // own stores visible before re-read
  const float inv = 1.0f / lsum;
  for (int j = lane; j < K; j += 32) {
    float s = outScore[base + j];
    outScore[base + j] = __expf(s - mx) * inv;
  }
}

extern "C" void kernel_launch(void* const* d_in, const int* in_sizes, int n_in,
                              void* d_out, int out_size, void* d_ws, size_t ws_size,
                              hipStream_t stream) {
  const float* q_c   = (const float*)d_in[0];
  const float* kv    = (const float*)d_in[1];
  const float* Wq    = (const float*)d_in[2];
  const float* bq    = (const float*)d_in[3];
  const float* Wk    = (const float*)d_in[4];
  const float* bk    = (const float*)d_in[5];
  const float* gamma = (const float*)d_in[6];
  const float* beta  = (const float*)d_in[7];
  const int*   cpr   = (const int*)d_in[8];
  (void)in_sizes; (void)n_in; (void)ws_size;

  char* ws = (char*)d_ws;
  unsigned short* qc_b = (unsigned short*)(ws + OFF_QC);
  unsigned short* kv_b = (unsigned short*)(ws + OFF_KV);
  unsigned short* wq_b = (unsigned short*)(ws + OFF_WQ);
  unsigned short* wk_b = (unsigned short*)(ws + OFF_WK);
  unsigned short* qi_b = (unsigned short*)(ws + OFF_QI);
  unsigned short* ki_b = (unsigned short*)(ws + OFF_KI);
  float*          scw  = (float*)(ws + OFF_SC);

  const int K = out_size / (2 * BSZ * SL);   // eff_k (indices + scores concatenated)

  const int NQC = BSZ * SL * QD, NKV = BSZ * SG * KVR, NWQ = DD * QD, NWK = DD * KVR;
  dsa_cvt_bf16<<<(NQC + 255) / 256, 256, 0, stream>>>(q_c, qc_b, NQC);
  dsa_cvt_bf16<<<(NKV + 255) / 256, 256, 0, stream>>>(kv, kv_b, NKV);
  dsa_cvt_bf16<<<(NWQ + 255) / 256, 256, 0, stream>>>(Wq, wq_b, NWQ);
  dsa_cvt_bf16<<<(NWK + 255) / 256, 256, 0, stream>>>(Wk, wk_b, NWK);

  dsa_kproj_ln<<<(BSZ * SG) / 16, 32, 0, stream>>>(kv_b, wk_b, bk, gamma, beta, ki_b);
  dsa_qproj  <<<(BSZ * SL) / 16, 32, 0, stream>>>(qc_b, wq_b, bq, qi_b);

  int* outIdx = (int*)d_out;
  float* outScore = (float*)d_out + (size_t)BSZ * SL * K;
  size_t dynShmem = 16 * 2 * 4096;   // TDM double buffers (ignored by fallback path)
  dsa_scores_topk<<<(BSZ * SL) / 16, 512, dynShmem, stream>>>(qi_b, ki_b, scw, cpr, K,
                                                              outIdx, outScore);
}
